// V1_PCBC_31705448579809
// MI455X (gfx1250) — matrix operand order repaired and checked by reading, not compiled, not run on hardware
//
#include <hip/hip_runtime.h>
#include <hip/hip_bf16.h>
#include <math.h>

typedef __attribute__((ext_vector_type(16))) _Float16 v16h;
typedef __attribute__((ext_vector_type(8)))  float    v8f;

#define EPS1 1e-4f
#define EPS2 50.f

// ---- CDNA5 async global->LDS copies (ASYNCcnt-tracked), CDNA5 ISA
__device__ __forceinline__ void async_copy_b64(unsigned lds_off, const void* g) {
    unsigned long long ga = (unsigned long long)(uintptr_t)g;
    asm volatile("global_load_async_to_lds_b64 %0, %1, off" :: "v"(lds_off), "v"(ga) : "memory");
}
__device__ __forceinline__ void wait_async() {
    asm volatile("s_wait_asynccnt 0x0" ::: "memory");
}

// ---------------- Stage A: RGB -> LAB (faithful to reference's odd where() math) ----
__global__ void k_rgb2lab(const float* __restrict__ x, float* __restrict__ lab) {
    int n = blockIdx.x * blockDim.x + threadIdx.x;      // 0 .. 262143 (B*H*W)
    int b = n >> 16; int hw = n & 65535;
    int base = b * 196608 + hw;                          // [B][3][H][W]
    float rgb[3] = { x[base], x[base + 65536], x[base + 131072] };
    float lin[3];
#pragma unroll
    for (int c = 0; c < 3; ++c) {
        float v = rgb[c];
        float g = (v < 0.0405f) ? 0.f : v;
        g = powf((g + 0.055f) / 1.055f, 2.4f) * 100.f;
        float l = ((v > 0.0405f) ? 0.f : v) * (100.f / 12.92f);
        lin[c] = g + l;
    }
    float X = (lin[0]*0.4124f + lin[1]*0.3576f + lin[2]*0.1805f) / 95.047f;
    float Y = (lin[0]*0.2126f + lin[1]*0.7152f + lin[2]*0.0722f) / 100.0f;
    float Z = (lin[0]*0.0193f + lin[1]*0.1192f + lin[2]*0.9505f) / 108.883f;
    float f[3]; float xyz[3] = { X, Y, Z };
#pragma unroll
    for (int c = 0; c < 3; ++c) {
        float t = xyz[c];
        float g = (t < 0.008856f) ? 0.f : t;
        g = cbrtf(fabsf(g) + 1e-5f);
        float l = 7.787f * ((t > 0.008856f) ? 0.f : t) + 16.f / 116.f;
        f[c] = g + l;
    }
    lab[base]          = (116.f * f[1] - 16.f) / 100.f;
    lab[base + 65536]  = (500.f * (f[0] - f[1]) + 128.f) / 255.f;
    lab[base + 131072] = (200.f * (f[1] - f[2]) + 128.f) / 255.f;
}

// ---------------- Stage A: 7x7 depthwise LoG conv + tanh + split ON/OFF ------------
__global__ void k_logconv(const float* __restrict__ lab, const float* __restrict__ filt,
                          float* __restrict__ xon, float* __restrict__ xoff) {
    int gx = blockIdx.x * 16 + threadIdx.x;
    int gy = blockIdx.y * 16 + threadIdx.y;
    int bc = blockIdx.z;                 // b*3 + c
    int c  = bc % 3;
    __shared__ float fs[49];
    int t = threadIdx.y * 16 + threadIdx.x;
    if (t < 49) fs[t] = filt[c * 49 + t];
    __syncthreads();
    const float* plane = lab + (bc << 16);
    float acc = 0.f;
#pragma unroll
    for (int i = 0; i < 7; ++i) {
        int yy = gy + i - 3;
        if ((unsigned)yy < 256u) {
#pragma unroll
            for (int j = 0; j < 7; ++j) {
                int xx = gx + j - 3;
                if ((unsigned)xx < 256u) acc = fmaf(fs[i*7 + j], plane[yy*256 + xx], acc);
            }
        }
    }
    float tv = tanhf(6.28318530718f * acc);
    int idx = (bc << 16) + gy * 256 + gx;
    xon[idx]  = fmaxf(tv, 0.f);
    xoff[idx] = fmaxf(-tv, 0.f);
}

// ---------------- Pack conv weights (32x6x9x9) into WMMA B layout, padded-K --------
// kp = row*16 + kx, row = c*9+ky (54 rows), kx 0..8 real, 9..15 zero-pad -> K = 864.
__global__ void k_pack(const float* __restrict__ on_sum, const float* __restrict__ off_sum,
                       _Float16* __restrict__ Bp) {
    int idx = blockIdx.x * 256 + threadIdx.x;            // 0 .. 27647
    if (idx >= 27648) return;
    int nt  = idx / 13824; int rem = idx - nt * 13824;   // 13824 = 27*32*16
    int s   = rem >> 9;
    int lane = (rem >> 4) & 31;
    int i   = rem & 15;
    int k   = 32 * s + ((i & 8) << 1) + (i & 7) + ((lane & 16) ? 8 : 0);
    int row = k >> 4, kx = k & 15;
    float v = 0.f;
    if (kx < 9) {
        int c = row / 9; int ky = row - 9 * c;
        int o = nt * 16 + (lane & 15);
        v = (c < 3) ? on_sum [((o*3 +  c     )*9 + ky)*9 + kx]
                    : off_sum[((o*3 + (c - 3))*9 + ky)*9 + kx];
    }
    Bp[idx] = (_Float16)v;
}

// ---------------- Pack norm weights (on_max/off_max) into B layout, N=0/1 used -----
// rows = kchan*9+dy (288), kx padded 9->16 -> K = 4608 = 144 steps of 32.
__global__ void k_packn(const float* __restrict__ on_max, const float* __restrict__ off_max,
                        _Float16* __restrict__ Bn) {
    int idx = blockIdx.x * 256 + threadIdx.x;            // 0 .. 73727
    if (idx >= 73728) return;
    int s    = idx >> 9;                                 // 0..143
    int lane = (idx >> 4) & 31;
    int i    = idx & 15;
    int k    = 32 * s + ((i & 8) << 1) + (i & 7) + ((lane & 16) ? 8 : 0);      // < 4608
    int row  = k >> 4, kx = k & 15;
    int n    = lane & 15;
    float v = 0.f;
    if (kx < 9 && n < 2) {
        int kchan = row / 9; int dy = row - 9 * kchan;
        v = (n == 0) ? on_max[kchan * 81 + dy * 9 + kx]
                     : off_max[kchan * 81 + dy * 9 + kx];
    }
    Bn[idx] = (_Float16)v;
}

__global__ void k_zero(float* __restrict__ p) {
    p[blockIdx.x * blockDim.x + threadIdx.x] = 0.f;
}

// ---------------- Per-iteration: norms as implicit GEMM on WMMA + e = x/norm -------
// Block: 128 threads = 4 waves; 64 pixels of one row; N-tile 16 (cols 0/1 = on/off).
__global__ __launch_bounds__(128) void k_norm_wmma(
        const _Float16* __restrict__ y16, const _Float16* __restrict__ Bn,
        const float* __restrict__ xon, const float* __restrict__ xoff,
        _Float16* __restrict__ e) {
    __shared__ _Float16 ysh[144 * 80];                   // 23,040 B: 16ch x 9dy rows, 80 cols
    __shared__ float norms[2][64];
    int tid = threadIdx.x;
    int w0 = blockIdx.x * 64, h = blockIdx.y, b = blockIdx.z;
    int lane = tid & 31, wv = tid >> 5;
    int hi8  = (lane >> 4) * 8;
    int mlow = lane & 15;
    int cb   = wv * 16 + mlow + hi8;
    bool interior = (blockIdx.x == 1 || blockIdx.x == 2) && (h >= 4) && (h <= 251);
    v8f acc = {};
    const v16h* bptr = (const v16h*)Bn;

    for (int half = 0; half < 2; ++half) {               // 16 channels per half
        __syncthreads();
        if (interior) {
            for (int l = tid; l < 2880; l += 128) {      // 144 rows x 20 b64 chunks
                int row = l / 20; int q = l - 20 * row;
                int kchan = half * 16 + row / 9; int dy = row % 9;
                int gy = h + dy - 4;
                unsigned loff = (unsigned)(row * 80 + q * 4) * 2u;   // 8B aligned
                async_copy_b64(loff, &y16[((b*32 + kchan) << 16) + gy*256 + (w0 - 4) + q*4]);
            }
            wait_async();
        } else {
            for (int l = tid; l < 11520; l += 128) {     // 144 x 80 halves
                int row = l / 80; int col = l - 80 * row;
                int kchan = half * 16 + row / 9; int dy = row % 9;
                int gy = h + dy - 4, gx = w0 - 4 + col;
                _Float16 v = (_Float16)0.f;
                if ((unsigned)gy < 256u && (unsigned)gx < 256u)
                    v = y16[((b*32 + kchan) << 16) + gy*256 + gx];
                ysh[l] = v;
            }
        }
        __syncthreads();
        for (int s = 0; s < 72; ++s) {                   // 72 steps x 2 halves = K 4608
            const _Float16* p0 = &ysh[(2 * s    ) * 80 + cb];
            const _Float16* p1 = &ysh[(2 * s + 1) * 80 + cb];
            v16h a;
#pragma unroll
            for (int q = 0; q < 8; ++q) { a[q] = p0[q]; a[8 + q] = p1[q]; }
            v16h bf = bptr[(half * 72 + s) * 32 + lane];
            acc = __builtin_amdgcn_wmma_f32_16x16x32_f16(false, a, false, bf, (short)0, acc, false, false);
        }
    }
    // D layout: lane = N, VGPR r = M(+8 hi). N=0 -> norm_on, N=1 -> norm_off.
    if (mlow < 2) {
        float* dst = norms[mlow];
#pragma unroll
        for (int r = 0; r < 8; ++r) dst[wv * 16 + hi8 + r] = acc[r];
    }
    __syncthreads();
    for (int l = tid; l < 384; l += 128) {               // 64 px x 6 e-channels
        int c6 = l / 64; int j = l - 64 * c6;
        float nrm = EPS2 + norms[(c6 < 3) ? 0 : 1][j];
        int cc = (c6 < 3) ? c6 : (c6 - 3);
        const float* xs = (c6 < 3) ? xon : xoff;
        float v = xs[((b * 3 + cc) << 16) + h * 256 + w0 + j] / nrm;
        e[((c6 * 4 + b) << 16) + h * 256 + w0 + j] = (_Float16)v;
    }
}

// ---------------- Per-iteration: implicit-GEMM conv via v_wmma_f32_16x16x32_f16 -----
// Plain f16 halo tile; A-fragment = 2 ds_load_b128 per K-step (compiler-fused).
// Also emits f16 copy of updated y (for next iteration's norm GEMM) unless final.
__global__ __launch_bounds__(128) void k_conv_wmma(
        const _Float16* __restrict__ e, const _Float16* __restrict__ Bp,
        const float* __restrict__ yold, float* __restrict__ ynew,
        _Float16* __restrict__ y16, int write16) {
    __shared__ _Float16 es[54 * 80];                     // 8,640 B: 6ch x 9ky rows, 80 cols
    int tid = threadIdx.x;
    int w0 = blockIdx.x * 64, h = blockIdx.y, b = blockIdx.z;
    bool interior = (blockIdx.x == 1 || blockIdx.x == 2) && (h >= 4) && (h <= 251);
    if (interior) {
        for (int l = tid; l < 1080; l += 128) {          // 54 rows x 20 b64 chunks
            int row = l / 20; int q = l - 20 * row;
            int c = row / 9;  int ky = row - 9 * c;
            int gy = h + ky - 4;
            unsigned loff = (unsigned)(row * 80 + q * 4) * 2u;       // 8B aligned
            async_copy_b64(loff, &e[((c * 4 + b) << 16) + gy*256 + (w0 - 4) + q*4]);
        }
        wait_async();
    } else {
        for (int l = tid; l < 4320; l += 128) {          // 54 x 80 halves
            int row = l / 80; int col = l - 80 * row;
            int c = row / 9;  int ky = row - 9 * c;
            int gy = h + ky - 4, gx = w0 - 4 + col;
            _Float16 v = (_Float16)0.f;
            if ((unsigned)gy < 256u && (unsigned)gx < 256u)
                v = e[((c * 4 + b) << 16) + gy * 256 + gx];
            es[l] = v;
        }
    }
    __syncthreads();

    int lane = tid & 31, wv = tid >> 5;
    int hi8  = (lane >> 4) * 8;
    int mlow = lane & 15;
    int cb   = wv * 16 + mlow + hi8;
    v8f acc0 = {}; v8f acc1 = {};
    const v16h* bptr = (const v16h*)Bp;

    for (int s = 0; s < 27; ++s) {                       // padded K = 864
        const _Float16* p0 = &es[(2 * s    ) * 80 + cb];
        const _Float16* p1 = &es[(2 * s + 1) * 80 + cb];
        v16h a;
#pragma unroll
        for (int q = 0; q < 8; ++q) { a[q] = p0[q]; a[8 + q] = p1[q]; }
        v16h b0 = bptr[(     s) * 32 + lane];
        v16h b1 = bptr[(27 + s) * 32 + lane];
        acc0 = __builtin_amdgcn_wmma_f32_16x16x32_f16(false, a, false, b0, (short)0, acc0, false, false);
        acc1 = __builtin_amdgcn_wmma_f32_16x16x32_f16(false, a, false, b1, (short)0, acc1, false, false);
    }

    int wbase = w0 + wv * 16 + hi8;
    int o0 = mlow, o1 = 16 + mlow;
#pragma unroll
    for (int r = 0; r < 8; ++r) {
        int i0 = ((b * 32 + o0) << 16) + h * 256 + wbase + r;
        float v0 = (EPS1 + yold[i0]) * acc0[r];
        ynew[i0] = v0;
        int i1 = ((b * 32 + o1) << 16) + h * 256 + wbase + r;
        float v1 = (EPS1 + yold[i1]) * acc1[r];
        ynew[i1] = v1;
        if (write16) { y16[i0] = (_Float16)v0; y16[i1] = (_Float16)v1; }
    }
}

extern "C" void kernel_launch(void* const* d_in, const int* in_sizes, int n_in,
                              void* d_out, int out_size, void* d_ws, size_t ws_size,
                              hipStream_t stream) {
    const float* x       = (const float*)d_in[0];
    const float* logf_   = (const float*)d_in[1];
    const float* on_sum  = (const float*)d_in[2];
    const float* off_sum = (const float*)d_in[3];
    const float* on_max  = (const float*)d_in[4];
    const float* off_max = (const float*)d_in[5];

    char* ws = (char*)d_ws;                              // ~60 MB total
    float*    lab  = (float*)   (ws + 0);                // 3,145,728 B
    float*    xon  = (float*)   (ws + 3145728);          // 3,145,728 B
    float*    xoff = (float*)   (ws + 6291456);          // 3,145,728 B
    _Float16* e    = (_Float16*)(ws + 9437184);          // 3,145,728 B
    _Float16* Bp   = (_Float16*)(ws + 12582912);         //    55,296 B
    _Float16* Bn   = (_Float16*)(ws + 12638208);         //   147,456 B
    _Float16* y16  = (_Float16*)(ws + 12785664);         // 16,777,216 B (f16 y, NCHW)
    float*    y    = (float*)   (ws + 29562880);         // 33,554,432 B (f32 y, NCHW)
    float*    outp = (float*)d_out;

    k_rgb2lab<<<dim3(1024),       dim3(256),    0, stream>>>(x, lab);
    k_logconv<<<dim3(16, 16, 12), dim3(16, 16), 0, stream>>>(lab, logf_, xon, xoff);
    k_pack   <<<dim3(108),        dim3(256),    0, stream>>>(on_sum, off_sum, Bp);
    k_packn  <<<dim3(288),        dim3(256),    0, stream>>>(on_max, off_max, Bn);
    k_zero   <<<dim3(32768),      dim3(256),    0, stream>>>(y);
    k_zero   <<<dim3(16384),      dim3(256),    0, stream>>>((float*)y16);

    for (int it = 0; it < 8; ++it) {
        k_norm_wmma<<<dim3(4, 256, 4), dim3(128), 0, stream>>>(y16, Bn, xon, xoff, e);
        float* tgt = (it == 7) ? outp : y;               // in-place y update; last -> d_out
        k_conv_wmma<<<dim3(4, 256, 4), dim3(128), 0, stream>>>(e, Bp, y, tgt, y16, (it < 7) ? 1 : 0);
    }
}
